// DeltaNetAttention_5299989643476
// MI455X (gfx1250) — matrix operand order repaired and checked by reading, not compiled, not run on hardware
//
#include <hip/hip_runtime.h>

// ---------------------------------------------------------------------------
// DeltaNet-style attention pipeline for gfx1250 (MI455X), bf16 WMMA everywhere.
//   B=8, L=256, D=2048, H=8, Dh=256  (Dh == L == 256)
// The causal linear-attention scan is algebraically collapsed into
//   Sc = tril(pq pk^T);  den = max(rowsum(Sc),1e-8);  out = (Sc @ V) / den
// so all heavy stages are GEMMs on v_wmma_f32_16x16x32_bf16.
// ---------------------------------------------------------------------------

typedef __bf16 bf16x8  __attribute__((ext_vector_type(8)));
typedef __bf16 bf16x16 __attribute__((ext_vector_type(16)));
typedef float  f32x8   __attribute__((ext_vector_type(8)));

#define DMODEL 2048
#define NB 8
#define NL 256
#define NH 8
#define NDH 256

enum { MF32 = 1, MBF = 2, MMASK = 4, MDEN = 8 };

__device__ __forceinline__ f32x8 wmma_bf16(bf16x16 a, bf16x16 b, f32x8 c) {
  return __builtin_amdgcn_wmma_f32_16x16x32_bf16(false, a, false, b,
                                                 (short)0, c, false, false);
}

// A-matrix 16x32 bf16 fragment (ISA 7.12.2 16-bit A layout):
// lane = {hi,r}: row m0+r; vector elems 0..7 hold K = hi*8..hi*8+7,
// elems 8..15 hold K = 16+hi*8 .. 23+hi*8.
__device__ __forceinline__ bf16x16 load_frag_A(const __bf16* __restrict__ base,
                                               int ld, int m0, int k0, int lane) {
  const int r  = lane & 15;
  const int hi = lane >> 4;
  const __bf16* p = base + (long long)(m0 + r) * ld + k0;
  union { bf16x16 v; bf16x8 h[2]; } u;
  u.h[0] = *(const bf16x8*)(p + hi * 8);
  u.h[1] = *(const bf16x8*)(p + 16 + hi * 8);
  return u.v;
}

// B-matrix 32x16 bf16 fragment: lane = {hi,c}: column n0+c (= row of Y),
// elems 0..7 hold K = hi*8..hi*8+7, elems 8..15 hold K = 16+hi*8 .. 23+hi*8.
__device__ __forceinline__ bf16x16 load_frag_B(const __bf16* __restrict__ base,
                                               int ld, int n0, int k0, int lane) {
  const int c  = lane & 15;
  const int hi = lane >> 4;
  const __bf16* p = base + (long long)(n0 + c) * ld + k0;
  union { bf16x16 v; bf16x8 h[2]; } u;
  u.h[0] = *(const bf16x8*)(p + hi * 8);
  u.h[1] = *(const bf16x8*)(p + 16 + hi * 8);
  return u.v;
}

// Generic batched C = X @ Y^T.  X:(M,K) lda, Y:(N,K) ldb, both bf16 row-major.
// Wave tile 32x64 (2 A-frags x 4 B-frags, 8 WMMAs/k-step); block = 8 waves
// arranged 2(M) x 4(N) -> 64x256 tile.  grid = (M/64, N/256, batch).
__global__ __launch_bounds__(256) void gemm_xyt(
    const __bf16* __restrict__ A, const __bf16* __restrict__ Bm,
    float* __restrict__ Cf, __bf16* __restrict__ Cb,
    const float* __restrict__ den,
    int lda, int ldb, int ldc, int K, int H,
    long long offAb, long long offAh,
    long long offBb, long long offBh,
    long long offCb, long long offCh,
    int denStride, int mode)
{
  const int z = blockIdx.z;
  const int b = z / H, h = z % H;
  A  += (long long)b * offAb + (long long)h * offAh;
  Bm += (long long)b * offBb + (long long)h * offBh;
  const long long coff = (long long)b * offCb + (long long)h * offCh;
  if (mode & MDEN) den += (long long)z * denStride;

  const int lane = threadIdx.x & 31;
  const int wave = threadIdx.x >> 5;
  const int wm = wave >> 2;          // 0..1
  const int wn = wave & 3;           // 0..3
  const int m0 = blockIdx.x * 64 + wm * 32;
  const int n0 = blockIdx.y * 256 + wn * 64;

  f32x8 acc[2][4] = {};
  for (int k0 = 0; k0 < K; k0 += 32) {
    bf16x16 a0 = load_frag_A(A, lda, m0,      k0, lane);
    bf16x16 a1 = load_frag_A(A, lda, m0 + 16, k0, lane);
#pragma unroll
    for (int j = 0; j < 4; ++j) {
      bf16x16 bfr = load_frag_B(Bm, ldb, n0 + j * 16, k0, lane);
      acc[0][j] = wmma_bf16(a0, bfr, acc[0][j]);
      acc[1][j] = wmma_bf16(a1, bfr, acc[1][j]);
    }
  }

  // C/D layout: lane&15 = N col, lane>>4 selects M rows r / r+8.
  const int cn = lane & 15;
  const int rs = (lane >> 4) * 8;
#pragma unroll
  for (int i = 0; i < 2; ++i) {
#pragma unroll
    for (int j = 0; j < 4; ++j) {
      const int n = n0 + j * 16 + cn;
#pragma unroll
      for (int r = 0; r < 8; ++r) {
        const int m = m0 + i * 16 + rs + r;
        float v = acc[i][j][r];
        if (mode & MMASK) v = (n <= m) ? v : 0.0f;       // causal, inclusive
        if (mode & MDEN)  v = v / den[m];
        const long long idx = coff + (long long)m * ldc + n;
        if (mode & MF32) Cf[idx] = v;
        if (mode & MBF)  Cb[idx] = (__bf16)v;
      }
    }
  }
}

// ---------------------------------------------------------------------------

__global__ __launch_bounds__(256) void cvt_bf16(const float* __restrict__ in,
                                                __bf16* __restrict__ out) {
  const long long i = (long long)blockIdx.x * 256 + threadIdx.x;
  out[i] = (__bf16)in[i];
}

// Stage 1 of the cs EMA: per-head partial sums of kv over (b,l).
// grid = (H, 16); block (h,c) sums 128 of the 2048 (b,l) rows for head h.
// Coalesced along m; the whole device participates instead of one WGP.
__global__ __launch_bounds__(256) void kv_mean_partial(
    const float* __restrict__ kv, float* __restrict__ partial) {
  const int h = blockIdx.x;        // 0..7
  const int c = blockIdx.y;        // 0..15
  const int m = threadIdx.x;       // 0..255
  float s = 0.0f;
  for (int r = 0; r < 128; ++r) {
    const int row = c * 128 + r;   // row = b*L + l
    const int b = row >> 8;
    const int l = row & 255;
    s += kv[(((long long)b * NH + h) * NL + l) * NL + m];
  }
  partial[((long long)h * 16 + c) * NDH + m] = s;
}

// Stage 2: fold 16 partials per head + tiny 8-step EMA over heads.
// cs[h] = carry BEFORE processing head h (matches jax.lax.scan ys).
__global__ __launch_bounds__(256) void cs_ema(const float* __restrict__ partial,
                                              float* __restrict__ cs,
                                              const float* __restrict__ alpha_p,
                                              const float* __restrict__ beta_p) {
  const int m = threadIdx.x;       // 0..255
  const float alpha = *alpha_p, beta = *beta_p;
  float c = 0.0f;
  for (int h = 0; h < NH; ++h) {
    cs[h * NDH + m] = c;
    float s = 0.0f;
#pragma unroll
    for (int j = 0; j < 16; ++j) s += partial[((long long)h * 16 + j) * NDH + m];
    c = beta * c + alpha * (s * (1.0f / (NB * NL)));
  }
}

__device__ __forceinline__ float phi1(float x) {
  return x > 0.0f ? x + 1.0f : __expf(x);    // elu(x)+1
}

// pq = phi(Q * (cs + alpha*(kv - cs))), pk = phi(K); outputs (B,H,L,Dh) bf16.
__global__ __launch_bounds__(256) void qmod_phi(
    const float* __restrict__ Qp, const float* __restrict__ Kp,
    const float* __restrict__ kv, const float* __restrict__ cs,
    __bf16* __restrict__ pq, __bf16* __restrict__ pk,
    const float* __restrict__ alpha_p) {
  const long long i = (long long)blockIdx.x * 256 + threadIdx.x;  // (b,h,l,d)
  const int d = (int)(i & 255);
  const int l = (int)((i >> 8) & 255);
  const int h = (int)((i >> 16) & 7);
  const int b = (int)(i >> 19);
  const float alpha = *alpha_p;
  const long long pidx = ((long long)(b * NL + l)) * DMODEL + h * NDH + d;
  const float q = Qp[pidx];
  const float k = Kp[pidx];
  const float c = cs[h * NDH + d];
  const float qm = q * (c + alpha * (kv[i] - c));
  pq[i] = (__bf16)phi1(qm);
  pk[i] = (__bf16)phi1(k);
}

// LDS-tiled transpose: Vt[(b,h,d,m)] = V[(b,m,h,d)].
// 64x64 tiles, padded LDS to avoid bank conflicts; both global accesses
// coalesced.  grid = (Dh/64, L/64, B*H), block = 256.
__global__ __launch_bounds__(256) void transpose_v(const __bf16* __restrict__ Vp,
                                                   __bf16* __restrict__ Vt) {
  __shared__ __bf16 tile[64][65];
  const int z = blockIdx.z;          // b*H + h
  const int b = z >> 3, h = z & 7;
  const int d0 = blockIdx.x * 64;
  const int m0 = blockIdx.y * 64;
  const int tx = threadIdx.x & 63;
  const int ty = threadIdx.x >> 6;   // 0..3
  const __bf16* src = Vp + ((long long)(b * NL + m0)) * DMODEL + h * NDH + d0;
#pragma unroll
  for (int r = 0; r < 64; r += 4)    // read in(m, d): coalesced along d
    tile[ty + r][tx] = src[(long long)(ty + r) * DMODEL + tx];
  __syncthreads();
  __bf16* dst = Vt + ((long long)z * NDH + d0) * NL + m0;
#pragma unroll
  for (int r = 0; r < 64; r += 4)    // write out(d, m): coalesced along m
    dst[(long long)(ty + r) * NL + tx] = tile[tx][ty + r];
}

// den[row] = max(rowsum(masked Sc), 1e-8); one wave per row, vector loads.
__global__ __launch_bounds__(256) void rowsum_den(const __bf16* __restrict__ Sc,
                                                  float* __restrict__ den) {
  const int lane = threadIdx.x & 31;
  const int wave = threadIdx.x >> 5;
  const long long row = (long long)blockIdx.x * 8 + wave;
  const bf16x8 v = *(const bf16x8*)(Sc + row * NL + lane * 8);
  float s = 0.0f;
#pragma unroll
  for (int i = 0; i < 8; ++i) s += (float)v[i];
#pragma unroll
  for (int o = 16; o; o >>= 1) s += __shfl_xor(s, o, 32);
  if (lane == 0) den[row] = fmaxf(s, 1e-8f);
}

// x = query + o + bo; LayerNorm(x) * g + b.  One block per row of D=2048.
__global__ __launch_bounds__(256) void resid_ln(
    const float* __restrict__ query, const float* __restrict__ o,
    const float* __restrict__ bo, const float* __restrict__ g,
    const float* __restrict__ bb, float* __restrict__ out) {
  __shared__ float red[256];
  const long long row = blockIdx.x;
  const int t = threadIdx.x;
  const float* q  = query + row * DMODEL;
  const float* ov = o     + row * DMODEL;
  float x[8];
  float s = 0.0f;
#pragma unroll
  for (int i = 0; i < 8; ++i) {
    const int c = t + i * 256;
    x[i] = q[c] + ov[c] + bo[c];
    s += x[i];
  }
  red[t] = s; __syncthreads();
  for (int st = 128; st > 0; st >>= 1) {
    if (t < st) red[t] += red[t + st];
    __syncthreads();
  }
  const float mu = red[0] * (1.0f / DMODEL);
  __syncthreads();
  float vs = 0.0f;
#pragma unroll
  for (int i = 0; i < 8; ++i) { const float dv = x[i] - mu; vs += dv * dv; }
  red[t] = vs; __syncthreads();
  for (int st = 128; st > 0; st >>= 1) {
    if (t < st) red[t] += red[t + st];
    __syncthreads();
  }
  const float inv = rsqrtf(red[0] * (1.0f / DMODEL) + 1e-5f);
#pragma unroll
  for (int i = 0; i < 8; ++i) {
    const int c = t + i * 256;
    out[row * DMODEL + c] = (x[i] - mu) * inv * g[c] + bb[c];
  }
}

// ---------------------------------------------------------------------------

extern "C" void kernel_launch(void* const* d_in, const int* in_sizes, int n_in,
                              void* d_out, int out_size, void* d_ws, size_t ws_size,
                              hipStream_t stream) {
  (void)in_sizes; (void)n_in; (void)out_size; (void)ws_size;
  const float* query = (const float*)d_in[0];
  const float* key_  = (const float*)d_in[1];
  const float* value = (const float*)d_in[2];
  const float* Wq = (const float*)d_in[3];
  const float* Wk = (const float*)d_in[4];
  const float* Wv = (const float*)d_in[5];
  const float* Wo = (const float*)d_in[6];
  const float* bo = (const float*)d_in[7];
  const float* ln_g = (const float*)d_in[8];
  const float* ln_b = (const float*)d_in[9];
  const float* alpha_p = (const float*)d_in[10];
  const float* beta_p  = (const float*)d_in[11];
  float* out = (float*)d_out;

  const long long N1 = (long long)NB * NL * DMODEL;   // 4,194,304
  const long long NW = (long long)DMODEL * DMODEL;    // 4,194,304

  char* w = (char*)d_ws;
  size_t off = 0;
  auto alloc = [&](size_t bytes) -> void* {
    void* p = w + off;
    off += (bytes + 255) & ~(size_t)255;
    return p;
  };
  __bf16* qbf   = (__bf16*)alloc(N1 * 2);
  __bf16* kbf   = (__bf16*)alloc(N1 * 2);
  __bf16* vbf   = (__bf16*)alloc(N1 * 2);
  __bf16* wqbf  = (__bf16*)alloc(NW * 2);
  __bf16* wkbf  = (__bf16*)alloc(NW * 2);
  __bf16* wvbf  = (__bf16*)alloc(NW * 2);
  __bf16* wobf  = (__bf16*)alloc(NW * 2);
  float*  Qp    = (float*) alloc(N1 * 4);
  float*  Kpf   = (float*) alloc(N1 * 4);
  __bf16* Kpb   = (__bf16*)alloc(N1 * 2);
  __bf16* Vpb   = (__bf16*)alloc(N1 * 2);
  float*  kvf   = (float*) alloc((long long)NB * NH * NL * NL * 4);
  float*  part  = (float*) alloc((long long)NH * 16 * NDH * 4);
  float*  cs    = (float*) alloc(NH * NDH * 4);
  __bf16* pqb   = (__bf16*)alloc(N1 * 2);
  __bf16* pkb   = (__bf16*)alloc(N1 * 2);
  __bf16* Vtb   = (__bf16*)alloc(N1 * 2);
  __bf16* Scb   = (__bf16*)alloc((long long)NB * NH * NL * NL * 2);
  float*  den   = (float*) alloc((long long)NB * NH * NL * 4);
  __bf16* attnb = (__bf16*)alloc(N1 * 2);
  float*  of    = (float*) alloc(N1 * 4);

  const int EB = (int)(N1 / 256);   // 16384 blocks for elementwise passes
  const dim3 blk(256);

  // 1) f32 -> bf16 conversions
  cvt_bf16<<<EB, blk, 0, stream>>>(query, qbf);
  cvt_bf16<<<EB, blk, 0, stream>>>(key_,  kbf);
  cvt_bf16<<<EB, blk, 0, stream>>>(value, vbf);
  cvt_bf16<<<(int)(NW / 256), blk, 0, stream>>>(Wq, wqbf);
  cvt_bf16<<<(int)(NW / 256), blk, 0, stream>>>(Wk, wkbf);
  cvt_bf16<<<(int)(NW / 256), blk, 0, stream>>>(Wv, wvbf);
  cvt_bf16<<<(int)(NW / 256), blk, 0, stream>>>(Wo, wobf);

  // 2) projections: (B*L,D) = X @ W^T, WMMA bf16
  const dim3 gBig(32, 8, 1);
  gemm_xyt<<<gBig, blk, 0, stream>>>(qbf, wqbf, Qp, nullptr, nullptr,
      DMODEL, DMODEL, DMODEL, DMODEL, 1, 0,0, 0,0, 0,0, 0, MF32);
  gemm_xyt<<<gBig, blk, 0, stream>>>(kbf, wkbf, Kpf, Kpb, nullptr,
      DMODEL, DMODEL, DMODEL, DMODEL, 1, 0,0, 0,0, 0,0, 0, MF32 | MBF);
  gemm_xyt<<<gBig, blk, 0, stream>>>(vbf, wvbf, nullptr, Vpb, nullptr,
      DMODEL, DMODEL, DMODEL, DMODEL, 1, 0,0, 0,0, 0,0, 0, MBF);

  // 3) kv[b,h] = K[b,h] @ V[b,h]^T  (batched over 64 heads)
  const dim3 gBat(4, 1, NB * NH);
  gemm_xyt<<<gBat, blk, 0, stream>>>(Kpb, Vpb, kvf, nullptr, nullptr,
      DMODEL, DMODEL, NL, NDH, NH,
      (long long)NL * DMODEL, NDH,
      (long long)NL * DMODEL, NDH,
      (long long)NH * NL * NL, (long long)NL * NL, 0, MF32);

  // 4) per-head EMA of kv mean (parallel partial reduction + tiny EMA)
  kv_mean_partial<<<dim3(NH, 16, 1), blk, 0, stream>>>(kvf, part);
  cs_ema<<<1, blk, 0, stream>>>(part, cs, alpha_p, beta_p);

  // 5) q_mod + phi -> pq, pk  (layout (B,H,L,Dh) bf16)
  qmod_phi<<<EB, blk, 0, stream>>>(Qp, Kpf, kvf, cs, pqb, pkb, alpha_p);

  // 6) V^T per head (LDS-tiled, fully coalesced)
  transpose_v<<<dim3(NDH / 64, NL / 64, NB * NH), blk, 0, stream>>>(Vpb, Vtb);

  // 7) Sc = tril(pq @ pk^T)  (collapsed causal linear-attention scan)
  gemm_xyt<<<gBat, blk, 0, stream>>>(pqb, pkb, nullptr, Scb, nullptr,
      NDH, NDH, NL, NDH, NH,
      (long long)NH * NL * NDH, (long long)NL * NDH,
      (long long)NH * NL * NDH, (long long)NL * NDH,
      (long long)NH * NL * NL, (long long)NL * NL, 0, MBF | MMASK);

  // 8) den = max(rowsum(Sc), 1e-8)
  rowsum_den<<<(NB * NH * NL) / 8, blk, 0, stream>>>(Scb, den);

  // 9) attn = (Sc @ V) / den, written straight into (B,L,H*Dh) bf16
  gemm_xyt<<<gBat, blk, 0, stream>>>(Scb, Vtb, nullptr, attnb, den,
      NL, NL, DMODEL, NL, NH,
      (long long)NH * NL * NL, (long long)NL * NL,
      (long long)NH * NDH * NL, (long long)NDH * NL,
      (long long)NL * DMODEL, NDH, NL, MBF | MDEN);

  // 10) o = attn @ Wo^T
  gemm_xyt<<<gBig, blk, 0, stream>>>(attnb, wobf, of, nullptr, nullptr,
      DMODEL, DMODEL, DMODEL, DMODEL, 1, 0,0, 0,0, 0,0, 0, MF32);

  // 11) residual + bias + LayerNorm -> out
  resid_ln<<<NB * NL, blk, 0, stream>>>(query, of, bo, ln_g, ln_b, out);
}
